// Former_31078383354489
// MI455X (gfx1250) — matrix operand order repaired and checked by reading, not compiled, not run on hardware
//
#include <hip/hip_runtime.h>
#include <hip/hip_bf16.h>

typedef __attribute__((ext_vector_type(16))) __bf16 v16bf;
typedef __attribute__((ext_vector_type(8)))  __bf16 v8bf;
typedef __attribute__((ext_vector_type(8)))  float  v8f;

#define DMODEL 1024
#define NHEAD  16
#define DHEAD  64
// permuted qkv layout strides (elements)
#define SECSZ  8388608   // 16 heads * 8192 rows * 64
#define HSZ    524288    // 8192 rows * 64

__device__ __forceinline__ __bf16 f2bf(float f) {
    unsigned u = __builtin_bit_cast(unsigned, f);
    u += 0x7FFFu + ((u >> 16) & 1u);           // round-to-nearest-even
    unsigned short hs = (unsigned short)(u >> 16);
    return __builtin_bit_cast(__bf16, hs);
}

__device__ __forceinline__ unsigned pack_bf2(float a, float b) {
    unsigned ua = __builtin_bit_cast(unsigned, a);
    ua += 0x7FFFu + ((ua >> 16) & 1u);
    unsigned ub = __builtin_bit_cast(unsigned, b);
    ub += 0x7FFFu + ((ub >> 16) & 1u);
    return (ua >> 16) | (ub & 0xFFFF0000u);
}

__device__ __forceinline__ v16bf concat8(v8bf lo, v8bf hi) {
    return __builtin_shufflevector(lo, hi, 0,1,2,3,4,5,6,7,8,9,10,11,12,13,14,15);
}

// CDNA5 async global->LDS copy (16B per lane), tracked by ASYNCcnt.
__device__ __forceinline__ void async_copy16(void* lds, const void* gptr) {
    unsigned laddr = (unsigned)(unsigned long long)lds;  // low 32b = LDS offset
    asm volatile("global_load_async_to_lds_b128 %0, %1, off"
                 :: "v"(laddr), "v"(gptr) : "memory");
}
__device__ __forceinline__ void wait_async0() {
    asm volatile("s_wait_asynccnt 0" ::: "memory");
}

// ---------------------------------------------------------------------------
// fp32 -> bf16 elementwise (n multiple of 1024); 4 elems/thread
// ---------------------------------------------------------------------------
__global__ __launch_bounds__(256)
void cvt_f32_bf16(const float* __restrict__ in, __bf16* __restrict__ out)
{
    long i = ((long)blockIdx.x * 256 + threadIdx.x) * 4;
    float4 t = *(const float4*)(in + i);
    uint2 r;
    r.x = pack_bf2(t.x, t.y);
    r.y = pack_bf2(t.z, t.w);
    *(uint2*)(out + i) = r;
}

// ---------------------------------------------------------------------------
// fp32 W[K][N] -> bf16 WT[N][K]   (grid: (N/32, K/32), 256 thr = 32x8)
// ---------------------------------------------------------------------------
__global__ __launch_bounds__(256)
void transpose_cvt(const float* __restrict__ W, __bf16* __restrict__ WT,
                   int K, int N)
{
    __shared__ float tile[32][33];
    int k0 = blockIdx.y * 32, n0 = blockIdx.x * 32;
    int tx = threadIdx.x & 31, ty = threadIdx.x >> 5;
    #pragma unroll
    for (int i = 0; i < 4; ++i)
        tile[ty + 8 * i][tx] = W[(long)(k0 + ty + 8 * i) * N + (n0 + tx)];
    __syncthreads();
    #pragma unroll
    for (int i = 0; i < 4; ++i)
        WT[(long)(n0 + ty + 8 * i) * K + (k0 + tx)] = f2bf(tile[tx][ty + 8 * i]);
}

// ---------------------------------------------------------------------------
// beff[n] = sum_k bin[k]*W[k,n] + badd[n]   (K = N = 1024, W fp32)
// ---------------------------------------------------------------------------
__global__ __launch_bounds__(256)
void bias_combine(const float* __restrict__ bin, const float* __restrict__ W,
                  const float* __restrict__ badd, float* __restrict__ beff)
{
    int n = blockIdx.x * blockDim.x + threadIdx.x;
    float s = 0.f;
    for (int kk = 0; kk < DMODEL; ++kk) s = fmaf(bin[kk], W[kk * DMODEL + n], s);
    beff[n] = s + badd[n];
}

// ---------------------------------------------------------------------------
// bf16 GEMM: C = act(A[M,K] @ W + bias), W TRANSPOSED: WT[N][K].
// Block 256 thr (8 waves), tile 128x128, k-step 32; wave tile 32x64 (8 WMMA).
// Double-buffered LDS + async global->LDS copies overlapped with compute:
//   wait(own copies, buf cur) -> barrier -> issue(buf next) -> WMMA(buf cur)
// cmode: 0 normal bf16 C[row][col]; 1 transposed CT[col][row] (weight compose);
//        2 qkv head-major permute [sec][h][row][64d].
// act: 0 none, 1 ReLU6.
// ---------------------------------------------------------------------------
__global__ __launch_bounds__(256)
void gemm_bf16(const __bf16* __restrict__ A, long lda,
               const __bf16* __restrict__ WT, long ldw,
               const float* __restrict__ bias,
               __bf16* __restrict__ C, long ldc,
               int K, int act, int cmode)
{
    __shared__ alignas(16) __bf16 sA[2][128][40];    // row stride 80B
    __shared__ alignas(16) __bf16 sBT[2][128][40];   // [col][k]

    const int tid  = threadIdx.x;
    const int wave = tid >> 5;
    const int lane = tid & 31;
    const int hl   = lane >> 4;
    const int l16  = lane & 15;

    const int m0 = blockIdx.y * 128;
    const int n0 = blockIdx.x * 128;
    const int wm = (wave >> 1) * 32;
    const int wn = (wave & 1) * 64;

    const v8f vzero = {0.f,0.f,0.f,0.f,0.f,0.f,0.f,0.f};
    v8f acc[2][4];
    #pragma unroll
    for (int i = 0; i < 2; ++i)
        #pragma unroll
        for (int j = 0; j < 4; ++j) acc[i][j] = vzero;

    auto issue = [&](int buf, int k0) {
        #pragma unroll
        for (int i = 0; i < 2; ++i) {        // A: 512 x 16B chunks
            int p = i * 256 + tid;
            int row = p >> 2, c8 = (p & 3) * 8;
            async_copy16(&sA[buf][row][c8], A + (long)(m0 + row) * lda + (k0 + c8));
        }
        #pragma unroll
        for (int i = 0; i < 2; ++i) {        // B from WT rows
            int p = i * 256 + tid;
            int col = p >> 2, c8 = (p & 3) * 8;
            async_copy16(&sBT[buf][col][c8], WT + (long)(n0 + col) * ldw + (k0 + c8));
        }
    };

    const int nsteps = K >> 5;
    issue(0, 0);
    for (int t = 0; t < nsteps; ++t) {
        wait_async0();        // my copies into buf cur complete
        __syncthreads();      // all waves' copies done; prior reads of next buf done
        if (t + 1 < nsteps) issue((t + 1) & 1, (t + 1) * 32);   // overlaps compute
        const int cur = t & 1;

        v16bf afrag[2], bfrag[4];
        #pragma unroll
        for (int s = 0; s < 2; ++s) {
            int row = wm + s * 16 + l16;
            afrag[s] = concat8(*(const v8bf*)&sA[cur][row][hl * 8],
                               *(const v8bf*)&sA[cur][row][16 + hl * 8]);
        }
        #pragma unroll
        for (int j = 0; j < 4; ++j) {
            int col = wn + j * 16 + l16;
            bfrag[j] = concat8(*(const v8bf*)&sBT[cur][col][hl * 8],
                               *(const v8bf*)&sBT[cur][col][16 + hl * 8]);
        }
        #pragma unroll
        for (int i = 0; i < 2; ++i)
            #pragma unroll
            for (int j = 0; j < 4; ++j)
                acc[i][j] = __builtin_amdgcn_wmma_f32_16x16x32_bf16(
                    false, afrag[i], false, bfrag[j], (short)0, acc[i][j],
                    false, false);
    }

    if (cmode == 1) {
        // CT[col][row]: 8 consecutive rows per lane -> one packed b128 store
        #pragma unroll
        for (int i = 0; i < 2; ++i)
            #pragma unroll
            for (int j = 0; j < 4; ++j) {
                int col  = n0 + wn + j * 16 + l16;
                int row0 = m0 + wm + i * 16 + hl * 8;
                uint4 r;
                r.x = pack_bf2(acc[i][j][0], acc[i][j][1]);
                r.y = pack_bf2(acc[i][j][2], acc[i][j][3]);
                r.z = pack_bf2(acc[i][j][4], acc[i][j][5]);
                r.w = pack_bf2(acc[i][j][6], acc[i][j][7]);
                *(uint4*)&C[(long)col * ldc + row0] = r;
            }
    } else if (cmode == 2) {
        // head-major qkv: col -> (sec, h, d); dst[row][d] contiguous per head
        #pragma unroll
        for (int i = 0; i < 2; ++i)
            #pragma unroll
            for (int j = 0; j < 4; ++j) {
                int base_row = m0 + wm + i * 16 + hl * 8;
                int col      = n0 + wn + j * 16 + l16;
                int sec = col >> 10, f = col & 1023;
                int hh = f & 15, dd = f >> 4;
                __bf16* dst = C + (long)sec * SECSZ + (long)hh * HSZ + dd;
                float bv = bias[col];
                #pragma unroll
                for (int e = 0; e < 8; ++e)
                    dst[(long)(base_row + e) * 64] = f2bf(acc[i][j][e] + bv);
            }
    } else {
        #pragma unroll
        for (int i = 0; i < 2; ++i)
            #pragma unroll
            for (int j = 0; j < 4; ++j) {
                int base_row = m0 + wm + i * 16 + hl * 8;
                int col      = n0 + wn + j * 16 + l16;
                float bv = bias ? bias[col] : 0.f;
                #pragma unroll
                for (int e = 0; e < 8; ++e) {
                    float val = acc[i][j][e] + bv;
                    if (act == 1) val = fminf(fmaxf(val, 0.f), 6.f);
                    C[(long)(base_row + e) * ldc + col] = f2bf(val);
                }
            }
    }
}

// ---------------------------------------------------------------------------
// Flash attention on head-major qkv: qkv[sec][h][row][64] (row = n*1024+tok).
// grid (8 qblk, 16 heads, 8 batch), 256 thr (8 waves x 16 q-rows).
// K tiles: async global->LDS (contiguous per token); V: transpose-on-store.
// Double-buffered, one barrier per k-step. Output vh[n,q,h*64+d] bf16.
// ---------------------------------------------------------------------------
__global__ __launch_bounds__(256)
void attention_flash(const __bf16* __restrict__ qkv, __bf16* __restrict__ vh)
{
    __shared__ alignas(16) __bf16 sK[2][32][72];     // [tok][d]
    __shared__ alignas(16) __bf16 sVT[2][64][40];    // [d][tok]
    __shared__ alignas(16) __bf16 sP[8][16][40];     // per-wave P scratch

    const int tid  = threadIdx.x;
    const int wave = tid >> 5;
    const int lane = tid & 31;
    const int hl   = lane >> 4;
    const int l16  = lane & 15;

    const int h  = blockIdx.y;
    const int nb = blockIdx.z;
    const int q0 = blockIdx.x * 128 + wave * 16;
    const long rbase = (long)nb * 1024;

    const __bf16* qp = qkv + (long)h * HSZ;
    const __bf16* kp = qkv + SECSZ + (long)h * HSZ;
    const __bf16* vp = qkv + 2 * (long)SECSZ + (long)h * HSZ;

    const v8f vzero = {0.f,0.f,0.f,0.f,0.f,0.f,0.f,0.f};

    // Q fragment: contiguous global v8bf loads (head-major layout)
    v16bf qa[2];
    #pragma unroll
    for (int s = 0; s < 2; ++s) {
        const __bf16* qr = qp + (rbase + q0 + l16) * 64 + s * 32;
        qa[s] = concat8(*(const v8bf*)(qr + hl * 8),
                        *(const v8bf*)(qr + 16 + hl * 8));
    }

    v8f O[4];
    O[0] = vzero; O[1] = vzero; O[2] = vzero; O[3] = vzero;
    float mrow[8], lrow[8];
    #pragma unroll
    for (int e = 0; e < 8; ++e) { mrow[e] = -1e30f; lrow[e] = 0.f; }

    const float scale = 1.0f / (8.0f + 1e-10f);

    auto stage = [&](int buf, int kt) {
        // K: 32 tok x 128B rows = 256 x 16B chunks, 1 per thread (async)
        {
            int tok = tid >> 3, c8 = (tid & 7) * 8;
            async_copy16(&sK[buf][tok][c8], kp + (rbase + kt + tok) * 64 + c8);
        }
        // V transposed: pairs over tok -> one ds_store_b32
        #pragma unroll
        for (int i = 0; i < 4; ++i) {
            int p  = i * 256 + tid;
            int dd = p & 63, tp = p >> 6;
            const __bf16* vr = vp + (rbase + kt + 2 * tp) * 64 + dd;
            unsigned lo = __builtin_bit_cast(unsigned short, vr[0]);
            unsigned hi = __builtin_bit_cast(unsigned short, vr[64]);
            ((unsigned*)&sVT[buf][dd][0])[tp] = lo | (hi << 16);
        }
    };

    stage(0, 0);
    for (int t = 0; t < 32; ++t) {
        wait_async0();
        __syncthreads();
        if (t + 1 < 32) stage((t + 1) & 1, (t + 1) * 32);   // overlaps compute
        const int cur = t & 1;

        // S = Q K^T : two 16x16 tiles
        v8f S[2];
        #pragma unroll
        for (int tt = 0; tt < 2; ++tt) {
            int tok = tt * 16 + l16;
            v16bf kb0 = concat8(*(const v8bf*)&sK[cur][tok][hl * 8],
                                *(const v8bf*)&sK[cur][tok][16 + hl * 8]);
            v16bf kb1 = concat8(*(const v8bf*)&sK[cur][tok][32 + hl * 8],
                                *(const v8bf*)&sK[cur][tok][48 + hl * 8]);
            v8f a = vzero;
            a = __builtin_amdgcn_wmma_f32_16x16x32_bf16(false, qa[0], false, kb0,
                                                        (short)0, a, false, false);
            a = __builtin_amdgcn_wmma_f32_16x16x32_bf16(false, qa[1], false, kb1,
                                                        (short)0, a, false, false);
            S[tt] = a;
        }

        // streaming softmax (row = 8*hl + e; 16-lane reductions)
        float alpha[8];
        #pragma unroll
        for (int e = 0; e < 8; ++e) {
            float s0 = S[0][e] * scale;
            float s1 = S[1][e] * scale;
            float mx = fmaxf(s0, s1);
            #pragma unroll
            for (int msk = 1; msk < 16; msk <<= 1)
                mx = fmaxf(mx, __shfl_xor(mx, msk, 32));
            float mnew = fmaxf(mrow[e], mx);
            alpha[e] = __expf(mrow[e] - mnew);
            mrow[e] = mnew;
            float p0 = __expf(s0 - mnew);
            float p1 = __expf(s1 - mnew);
            S[0][e] = p0; S[1][e] = p1;
            float rs = p0 + p1;
            #pragma unroll
            for (int msk = 1; msk < 16; msk <<= 1)
                rs += __shfl_xor(rs, msk, 32);
            lrow[e] = lrow[e] * alpha[e] + rs;
        }
        #pragma unroll
        for (int tt = 0; tt < 4; ++tt)
            #pragma unroll
            for (int e = 0; e < 8; ++e) O[tt][e] *= alpha[e];

        // relay P (wave-private; same-wave LDS ops are in-order)
        #pragma unroll
        for (int e = 0; e < 8; ++e) {
            sP[wave][8 * hl + e][l16]      = f2bf(S[0][e]);
            sP[wave][8 * hl + e][16 + l16] = f2bf(S[1][e]);
        }
        v16bf pa = concat8(*(const v8bf*)&sP[wave][l16][hl * 8],
                           *(const v8bf*)&sP[wave][l16][16 + hl * 8]);

        #pragma unroll
        for (int tt = 0; tt < 4; ++tt) {
            int dcol = tt * 16 + l16;
            v16bf vb = concat8(*(const v8bf*)&sVT[cur][dcol][hl * 8],
                               *(const v8bf*)&sVT[cur][dcol][16 + hl * 8]);
            O[tt] = __builtin_amdgcn_wmma_f32_16x16x32_bf16(false, pa, false, vb,
                                                            (short)0, O[tt],
                                                            false, false);
        }
    }

    #pragma unroll
    for (int t = 0; t < 4; ++t) {
        int dcol = t * 16 + l16;
        #pragma unroll
        for (int e = 0; e < 8; ++e) {
            int qrow = q0 + 8 * hl + e;
            vh[(rbase + qrow) * DMODEL + h * DHEAD + dcol] =
                f2bf(O[t][e] / lrow[e]);
        }
    }
}

// ---------------------------------------------------------------------------
// LayerNorm over 1024 features (bf16 in). Two variants: bf16 out / fp32 out.
// ---------------------------------------------------------------------------
__device__ __forceinline__ void ln_body(const __bf16* x, float* loc,
                                        float& mu, float& inv, float* red, int t)
{
    float s = 0.f;
    #pragma unroll
    for (int i = 0; i < 4; ++i) { loc[i] = (float)x[t + 256 * i]; s += loc[i]; }
    #pragma unroll
    for (int msk = 1; msk < 32; msk <<= 1) s += __shfl_xor(s, msk, 32);
    if ((t & 31) == 0) red[t >> 5] = s;
    __syncthreads();
    float tot = 0.f;
    #pragma unroll
    for (int w = 0; w < 8; ++w) tot += red[w];
    mu = tot * (1.0f / 1024.0f);
    __syncthreads();
    float vs = 0.f;
    #pragma unroll
    for (int i = 0; i < 4; ++i) { float d = loc[i] - mu; vs += d * d; }
    #pragma unroll
    for (int msk = 1; msk < 32; msk <<= 1) vs += __shfl_xor(vs, msk, 32);
    if ((t & 31) == 0) red[t >> 5] = vs;
    __syncthreads();
    float vt = 0.f;
    #pragma unroll
    for (int w = 0; w < 8; ++w) vt += red[w];
    inv = rsqrtf(vt * (1.0f / 1024.0f) + 1e-5f);
}

__global__ __launch_bounds__(256)
void layernorm_bf(const __bf16* __restrict__ in, __bf16* __restrict__ out,
                  const float* __restrict__ g, const float* __restrict__ b)
{
    __shared__ float red[8];
    long row = blockIdx.x;
    int t = threadIdx.x;
    float loc[4], mu, inv;
    ln_body(in + row * DMODEL, loc, mu, inv, red, t);
    #pragma unroll
    for (int i = 0; i < 4; ++i) {
        int col = t + 256 * i;
        out[row * DMODEL + col] = f2bf((loc[i] - mu) * inv * g[col] + b[col]);
    }
}

__global__ __launch_bounds__(256)
void layernorm_f32(const __bf16* __restrict__ in, float* __restrict__ out,
                   const float* __restrict__ g, const float* __restrict__ b)
{
    __shared__ float red[8];
    long row = blockIdx.x;
    int t = threadIdx.x;
    float loc[4], mu, inv;
    ln_body(in + row * DMODEL, loc, mu, inv, red, t);
    #pragma unroll
    for (int i = 0; i < 4; ++i) {
        int col = t + 256 * i;
        out[row * DMODEL + col] = (loc[i] - mu) * inv * g[col] + b[col];
    }
}

// ---------------------------------------------------------------------------
extern "C" void kernel_launch(void* const* d_in, const int* in_sizes, int n_in,
                              void* d_out, int out_size, void* d_ws, size_t ws_size,
                              hipStream_t stream)
{
    (void)in_sizes; (void)n_in; (void)out_size; (void)ws_size;

    const float* x     = (const float*)d_in[0];
    const float* w_qvk = (const float*)d_in[1];
    const float* b_qvk = (const float*)d_in[2];
    const float* w_q   = (const float*)d_in[3];
    const float* b_q   = (const float*)d_in[4];
    const float* w_k   = (const float*)d_in[5];
    const float* b_k   = (const float*)d_in[6];
    const float* w_v   = (const float*)d_in[7];
    const float* b_v   = (const float*)d_in[8];
    const float* w_o   = (const float*)d_in[9];
    const float* b_o   = (const float*)d_in[10];
    const float* ln1g  = (const float*)d_in[11];
    const float* ln1b  = (const float*)d_in[12];
    const float* w1    = (const float*)d_in[13];
    const float* b1    = (const float*)d_in[14];
    const float* w2    = (const float*)d_in[15];
    const float* b2    = (const float*)d_in[16];
    const float* ln2g  = (const float*)d_in[17];
    const float* ln2b  = (const float*)d_in[18];
    float* out = (float*)d_out;

    // workspace (byte offsets; all 16B aligned); peak ~140 MB
    char* wsb = (char*)d_ws;
    __bf16* xb     = (__bf16*)(wsb + 0);            // 8192x1024
    __bf16* qkv    = (__bf16*)(wsb + 16777216);     // head-major, 3*16*8192*64
    __bf16* vhb    = (__bf16*)(wsb + 67108864);     // 8192x1024
    __bf16* hid    = (__bf16*)(wsb + 83886080);     // 8192x2048
    __bf16* attb   = (__bf16*)(wsb + 16777216);     // reuse qkv lo
    __bf16* outpre = (__bf16*)(wsb + 33554432);     // reuse qkv mid
    __bf16* wqvkb  = (__bf16*)(wsb + 117440512);    // 1024x3072
    __bf16* wtq    = (__bf16*)(wsb + 123731968);    // 1024x1024 (WT)
    __bf16* wtk    = (__bf16*)(wsb + 125829120);
    __bf16* wtv    = (__bf16*)(wsb + 127926272);
    __bf16* wtall  = (__bf16*)(wsb + 130023424);    // 3072x1024 composed WT
    __bf16* wto    = (__bf16*)(wsb + 136314880);    // 1024x1024
    __bf16* wt1    = (__bf16*)(wsb + 138412032);    // 2048x1024
    __bf16* wt2    = (__bf16*)(wsb + 142606336);    // 1024x2048
    float*  beff   = (float*)(wsb + 146800640);     // 3072 fp32

    dim3 blk(256);
    const int M = 8 * 1024;

    // 0) one-time bf16 conversion / weight transposition
    cvt_f32_bf16<<<8192, blk, 0, stream>>>(x, xb);
    cvt_f32_bf16<<<3072, blk, 0, stream>>>(w_qvk, wqvkb);
    transpose_cvt<<<dim3(32, 32), blk, 0, stream>>>(w_q, wtq, 1024, 1024);
    transpose_cvt<<<dim3(32, 32), blk, 0, stream>>>(w_k, wtk, 1024, 1024);
    transpose_cvt<<<dim3(32, 32), blk, 0, stream>>>(w_v, wtv, 1024, 1024);
    transpose_cvt<<<dim3(32, 32), blk, 0, stream>>>(w_o, wto, 1024, 1024);
    transpose_cvt<<<dim3(64, 32), blk, 0, stream>>>(w1, wt1, 1024, 2048);
    transpose_cvt<<<dim3(32, 64), blk, 0, stream>>>(w2, wt2, 2048, 1024);
    bias_combine<<<4, blk, 0, stream>>>(b_qvk,        w_q, b_q, beff);
    bias_combine<<<4, blk, 0, stream>>>(b_qvk + 1024, w_k, b_k, beff + 1024);
    bias_combine<<<4, blk, 0, stream>>>(b_qvk + 2048, w_v, b_v, beff + 2048);

    // 1) compose projection weights directly into WT layout (cmode=1)
    gemm_bf16<<<dim3(8, 8), blk, 0, stream>>>(wqvkb,        3072, wtq, 1024, nullptr, wtall,               1024, 1024, 0, 1);
    gemm_bf16<<<dim3(8, 8), blk, 0, stream>>>(wqvkb + 1024, 3072, wtk, 1024, nullptr, wtall + 1024 * 1024, 1024, 1024, 0, 1);
    gemm_bf16<<<dim3(8, 8), blk, 0, stream>>>(wqvkb + 2048, 3072, wtv, 1024, nullptr, wtall + 2048 * 1024, 1024, 1024, 0, 1);

    // 2) fused qkv = x @ Weff_all + beff, written head-major (cmode=2)
    gemm_bf16<<<dim3(24, M / 128), blk, 0, stream>>>(xb, 1024, wtall, 1024, beff, qkv, 0, 1024, 0, 2);

    // 3) flash attention on head-major qkv
    attention_flash<<<dim3(8, 16, 8), blk, 0, stream>>>(qkv, vhb);

    // 4) output projection + LN1 (in place, bf16)
    gemm_bf16<<<dim3(8, M / 128), blk, 0, stream>>>(vhb, 1024, wto, 1024, b_o, attb, 1024, 1024, 0, 0);
    layernorm_bf<<<M, blk, 0, stream>>>(attb, attb, ln1g, ln1b);

    // 5) MLP
    gemm_bf16<<<dim3(16, M / 128), blk, 0, stream>>>(attb, 1024, wt1, 1024, b1, hid, 2048, 1024, 1, 0);
    gemm_bf16<<<dim3(8,  M / 128), blk, 0, stream>>>(hid, 2048, wt2, 2048, b2, outpre, 1024, 2048, 0, 0);

    // 6) LN2 -> fp32 output
    layernorm_f32<<<M, blk, 0, stream>>>(outpre, out, ln2g, ln2b);
}
